// DiTBlockLite_63067299775130
// MI455X (gfx1250) — compile-verified, matrix operand order checked by reading, not run on hardware
//
#include <hip/hip_runtime.h>
#include <cstdint>

// ---------------------------------------------------------------------------
// CDNA5 (gfx1250, wave32) DiT block: LN -> QKV GEMM -> flash attention ->
// proj GEMM (+resid) -> LN -> FC1+SiLU -> FC2 (+resid).
// Matmuls through v_wmma_f32_16x16x32_bf16; operand staging through the
// Tensor Data Mover (tensor_load_to_lds + s_wait_tensorcnt), double-buffered;
// softmax reductions via v_permlane16; L2 warmed with global_prefetch_b8.
// ---------------------------------------------------------------------------

typedef __attribute__((ext_vector_type(16))) __bf16        v16bf;
typedef __attribute__((ext_vector_type(8)))  float         v8f;
typedef __attribute__((ext_vector_type(4)))  unsigned int  v4u;
typedef __attribute__((ext_vector_type(8)))  int           v8i;
typedef __attribute__((ext_vector_type(4)))  int           v4i;

#if __has_builtin(__builtin_amdgcn_tensor_load_to_lds) && \
    __has_builtin(__builtin_amdgcn_s_wait_tensorcnt)
#define HAVE_TDM 1
#else
#define HAVE_TDM 0
#endif

#if __has_builtin(__builtin_amdgcn_permlane16)
#define HAVE_PERMLANE 1
#else
#define HAVE_PERMLANE 0
#endif

__device__ __forceinline__ v8f wmma_bf16(v16bf a, v16bf b, v8f c) {
  return __builtin_amdgcn_wmma_f32_16x16x32_bf16(
      false, a, false, b, (short)0, c, false, false);
}

__device__ __forceinline__ __bf16 f2bf(float f) { return (__bf16)f; }

// 16-lane-row butterfly reductions via v_permlane16 (no LDS, no dscnt waits).
#if HAVE_PERMLANE
__device__ __forceinline__ float permx(float v, unsigned s1, unsigned s2) {
  int i = __builtin_bit_cast(int, v);
  i = __builtin_amdgcn_permlane16(i, i, s1, s2, false, false);
  return __builtin_bit_cast(float, i);
}
__device__ __forceinline__ float row16_max(float v) {
  v = fmaxf(v, permx(v, 0x67452301u, 0xEFCDAB89u));  // xor 1
  v = fmaxf(v, permx(v, 0x54761032u, 0xDCFE98BAu));  // xor 2
  v = fmaxf(v, permx(v, 0x32107654u, 0xBA98FEDCu));  // xor 4
  v = fmaxf(v, permx(v, 0xFEDCBA98u, 0x76543210u));  // xor 8
  return v;
}
__device__ __forceinline__ float row16_sum(float v) {
  v += permx(v, 0x67452301u, 0xEFCDAB89u);
  v += permx(v, 0x54761032u, 0xDCFE98BAu);
  v += permx(v, 0x32107654u, 0xBA98FEDCu);
  v += permx(v, 0xFEDCBA98u, 0x76543210u);
  return v;
}
#else
__device__ __forceinline__ float row16_max(float v) {
  for (int off = 1; off < 16; off <<= 1) v = fmaxf(v, __shfl_xor(v, off, 32));
  return v;
}
__device__ __forceinline__ float row16_sum(float v) {
  for (int off = 1; off < 16; off <<= 1) v += __shfl_xor(v, off, 32);
  return v;
}
#endif

// WMMA 16x32 bf16 operand fragment (ISA 7.12.2 A/B lane layout):
// lanes 0-15 = row (lane&15), K {0..7,16..23}; lanes 16-31 same rows,
// K {8..15,24..31}.  `stride` in elements.
__device__ __forceinline__ v16bf load_frag(const __bf16* p0, int stride) {
  int lane = threadIdx.x & 31;
  const __bf16* p = p0 + (size_t)(lane & 15) * stride + ((lane >> 4) << 3);
  union { v16bf v; uint4 q[2]; } u;
  u.q[0] = *(const uint4*)(p);
  u.q[1] = *(const uint4*)(p + 16);
  return u.v;
}

#if HAVE_TDM
// One TDM 2D tile load: tile_rows x tile_k bf16, global row stride
// `row_stride` elements, into LDS at byte offset `lds_off`.  TDM pad feature
// inserts `pad_amt+1` DWORDs after every 2^(pad_int+1) data DWORDs so LDS
// rows land on our bank-conflict-free strides.
__device__ __forceinline__ void tdm_load_tile(
    unsigned lds_off, const __bf16* gptr, unsigned tile_k, unsigned tile_rows,
    unsigned tensor_k, unsigned row_stride, unsigned pad_int, unsigned pad_amt) {
  unsigned long long ga = (unsigned long long)(uintptr_t)gptr;
  v4u g0;
  g0[0] = 1u;                                            // count=1, user mode
  g0[1] = lds_off;                                       // lds_addr [63:32]
  g0[2] = (unsigned)(ga & 0xFFFFFFFFu);                  // global_addr lo
  g0[3] = (unsigned)((ga >> 32) & 0x01FFFFFFu) | (2u << 30);  // addr hi | type=2
  v8i g1;
  g1[0] = (1 << 16)                                      // data_size = 2 bytes
        | (1 << 20)                                      // pad_enable
        | ((int)pad_int << 22)                           // pad_interval
        | ((int)pad_amt << 25);                          // pad_amount
  g1[1] = (int)((tensor_k & 0xFFFFu) << 16);             // tensor_dim0 lo16
  g1[2] = (int)((tensor_k >> 16) & 0xFFFFu);             // dim0 hi16; dim1 lo16=0
  g1[3] = (int)(16u)                                     // tensor_dim1 = 1<<20
        | (int)((tile_k & 0xFFFFu) << 16);               // tile_dim0
  g1[4] = (int)(tile_rows & 0xFFFFu);                    // tile_dim1 (dim2=0)
  g1[5] = (int)row_stride;                               // tensor_dim0_stride lo32
  g1[6] = 0;
  g1[7] = 0;
  v4i z = {0, 0, 0, 0};
#if __clang_major__ >= 23
  v8i z8 = {0, 0, 0, 0, 0, 0, 0, 0};
  __builtin_amdgcn_tensor_load_to_lds(g0, g1, z, z, z8, 0);
#else
  __builtin_amdgcn_tensor_load_to_lds(g0, g1, z, z, 0);
#endif
}
#endif  // HAVE_TDM

// ---------------------------------------------------------------------------
// Weight transpose + fp32 -> bf16 cast:  Wt[n][k] = bf16(W[k][n])
// ---------------------------------------------------------------------------
__global__ __launch_bounds__(256) void transpose_cast_kernel(
    const float* __restrict__ W, __bf16* __restrict__ Wt, int K, int N) {
  __shared__ float tile[32][33];
  int k0 = blockIdx.x * 32, n0 = blockIdx.y * 32;
  int tx = threadIdx.x & 31, ty = threadIdx.x >> 5;
#pragma unroll
  for (int i = ty; i < 32; i += 8)
    tile[i][tx] = W[(size_t)(k0 + i) * N + (n0 + tx)];
  __syncthreads();
#pragma unroll
  for (int i = ty; i < 32; i += 8)
    Wt[(size_t)(n0 + i) * K + (k0 + tx)] = f2bf(tile[tx][i]);
}

// ---------------------------------------------------------------------------
// LayerNorm (row = 1024 fp32) -> bf16.
// ---------------------------------------------------------------------------
__global__ __launch_bounds__(256) void layernorm_bf16_kernel(
    const float* __restrict__ x, const float* __restrict__ w,
    const float* __restrict__ b, __bf16* __restrict__ out, int D) {
  int row = blockIdx.x;
  int t = threadIdx.x;
  const float4 xv = ((const float4*)(x + (size_t)row * D))[t];
  float s  = xv.x + xv.y + xv.z + xv.w;
  float s2 = xv.x * xv.x + xv.y * xv.y + xv.z * xv.z + xv.w * xv.w;
#pragma unroll
  for (int off = 16; off > 0; off >>= 1) {
    s  += __shfl_xor(s,  off, 32);
    s2 += __shfl_xor(s2, off, 32);
  }
  __shared__ float red[2][8];
  if ((t & 31) == 0) { red[0][t >> 5] = s; red[1][t >> 5] = s2; }
  __syncthreads();
  float ts = 0.f, ts2 = 0.f;
#pragma unroll
  for (int i = 0; i < 8; ++i) { ts += red[0][i]; ts2 += red[1][i]; }
  float inv = 1.0f / (float)D;
  float mu  = ts * inv;
  float var = ts2 * inv - mu * mu;
  float rs  = rsqrtf(var + 1e-6f);
  int base = t * 4;
  union { __bf16 e[4]; uint2 q; } o;
  o.e[0] = f2bf((xv.x - mu) * rs * w[base + 0] + b[base + 0]);
  o.e[1] = f2bf((xv.y - mu) * rs * w[base + 1] + b[base + 1]);
  o.e[2] = f2bf((xv.z - mu) * rs * w[base + 2] + b[base + 2]);
  o.e[3] = f2bf((xv.w - mu) * rs * w[base + 3] + b[base + 3]);
  *(uint2*)(out + (size_t)row * D + base) = o.q;
}

// ---------------------------------------------------------------------------
// Tiled WMMA GEMM:  C[M,N] = A[M,K] * Bt[N,K]^T  (+bias, +SiLU, +resid)
// Block tile 128x128, 4 waves, each wave 64x64 via 4x4 WMMA.  A/B tiles
// staged by TDM, double-buffered; L2 warmed 4 K-steps ahead by prefetch.
// LDS rows padded to 40 bf16 (80B) by the TDM pad feature.
// ---------------------------------------------------------------------------
#define LDSK 40

template <int ACT, bool RESID, bool OUTBF16>
__global__ __launch_bounds__(128) void gemm_wmma_kernel(
    const __bf16* __restrict__ A, const __bf16* __restrict__ Bt,
    const float* __restrict__ bias, const float* __restrict__ resid,
    void* __restrict__ out, int M, int N, int K) {
  __shared__ __bf16 lsA[2][128][LDSK];
  __shared__ __bf16 lsB[2][128][LDSK];

  const int bm = blockIdx.x * 128;
  const int bn = blockIdx.y * 128;
  const int tid  = threadIdx.x;
  const int wave = tid >> 5;
  const int lane = tid & 31;
  const int wm = (wave & 1) * 64;
  const int wn = (wave >> 1) * 64;

  v8f acc[4][4] = {};
  const int nk = K / 32;

#if HAVE_TDM
  const unsigned offA0 = (unsigned)(uintptr_t)(&lsA[0][0][0]);
  const unsigned offA1 = (unsigned)(uintptr_t)(&lsA[1][0][0]);
  const unsigned offB0 = (unsigned)(uintptr_t)(&lsB[0][0][0]);
  const unsigned offB1 = (unsigned)(uintptr_t)(&lsB[1][0][0]);
  if (wave == 0) {
    tdm_load_tile(offA0, A  + (size_t)bm * K, 32, 128, (unsigned)K, (unsigned)K, 3, 3);
    tdm_load_tile(offB0, Bt + (size_t)bn * K, 32, 128, (unsigned)K, (unsigned)K, 3, 3);
  }
  for (int it = 0; it < nk; ++it) {
    const int cur = it & 1;
    if (wave == 0) __builtin_amdgcn_s_wait_tensorcnt(0);
    __syncthreads();                         // staged tile visible to all waves
    if (wave == 0 && (it + 1) < nk) {        // prefetch next tile into alt buf
      const int kt = (it + 1) * 32;
      tdm_load_tile(cur ? offA0 : offA1, A  + (size_t)bm * K + kt, 32, 128,
                    (unsigned)K, (unsigned)K, 3, 3);
      tdm_load_tile(cur ? offB0 : offB1, Bt + (size_t)bn * K + kt, 32, 128,
                    (unsigned)K, (unsigned)K, 3, 3);
    }
    if (it + 4 < nk) {                       // warm L2 ahead of the TDM
      __builtin_prefetch(A  + (size_t)(bm + tid) * K + (it + 4) * 32, 0, 0);
      __builtin_prefetch(Bt + (size_t)(bn + tid) * K + (it + 4) * 32, 0, 0);
    }
    v16bf bfrag[4];
#pragma unroll
    for (int j = 0; j < 4; ++j)
      bfrag[j] = load_frag(&lsB[cur][wn + j * 16][0], LDSK);
#pragma unroll
    for (int i = 0; i < 4; ++i) {
      v16bf afrag = load_frag(&lsA[cur][wm + i * 16][0], LDSK);
#pragma unroll
      for (int j = 0; j < 4; ++j)
        acc[i][j] = wmma_bf16(afrag, bfrag[j], acc[i][j]);
    }
    __syncthreads();                         // done reading before overwrite
  }
#else
  for (int it = 0; it < nk; ++it) {
    const int cur = it & 1;
    const int kt = it * 32;
    {
      const uint4* ga = (const uint4*)(A  + (size_t)(bm + tid) * K + kt);
      const uint4* gb = (const uint4*)(Bt + (size_t)(bn + tid) * K + kt);
      uint4* la = (uint4*)(&lsA[cur][tid][0]);
      uint4* lb = (uint4*)(&lsB[cur][tid][0]);
#pragma unroll
      for (int v = 0; v < 4; ++v) { la[v] = ga[v]; lb[v] = gb[v]; }
    }
    __syncthreads();
    v16bf bfrag[4];
#pragma unroll
    for (int j = 0; j < 4; ++j)
      bfrag[j] = load_frag(&lsB[cur][wn + j * 16][0], LDSK);
#pragma unroll
    for (int i = 0; i < 4; ++i) {
      v16bf afrag = load_frag(&lsA[cur][wm + i * 16][0], LDSK);
#pragma unroll
      for (int j = 0; j < 4; ++j)
        acc[i][j] = wmma_bf16(afrag, bfrag[j], acc[i][j]);
    }
    __syncthreads();
  }
#endif

  // Epilogue: reg r -> row r (lanes 0-15) / r+8 (lanes 16-31); col = lane&15.
  const int nloc  = lane & 15;
  const int rbase = (lane >> 4) * 8;
#pragma unroll
  for (int i = 0; i < 4; ++i) {
#pragma unroll
    for (int j = 0; j < 4; ++j) {
      const int col = bn + wn + j * 16 + nloc;
      const float bv = bias[col];
#pragma unroll
      for (int r = 0; r < 8; ++r) {
        const int row = bm + wm + i * 16 + rbase + r;
        float c = acc[i][j][r] + bv;
        if (ACT == 1) c = c * (1.0f / (1.0f + __expf(-c)));  // SiLU
        if (RESID) c += resid[(size_t)row * N + col];
        if constexpr (OUTBF16)
          ((__bf16*)out)[(size_t)row * N + col] = f2bf(c);
        else
          ((float*)out)[(size_t)row * N + col] = c;
      }
    }
  }
}

// ---------------------------------------------------------------------------
// Flash attention over qkv [M, 3072] bf16 (Q|K|V), 16 heads, hd=64, seq 2048.
// Block = 4 waves = 64 queries of one (batch, head).  K tiles staged once per
// block by TDM (double-buffered); V tiles transposed into LDS with packed
// b64 stores; online softmax with permlane16 row reductions.
// ---------------------------------------------------------------------------
__global__ __launch_bounds__(128) void flash_attn_kernel(
    const __bf16* __restrict__ qkv, __bf16* __restrict__ out, int Nseq) {
  const int ROWS = 3072;
  const int bh = blockIdx.y;
  const int b = bh >> 4, h = bh & 15;
  const int tid = threadIdx.x, wave = tid >> 5, lane = tid & 31;
  const int qrow = blockIdx.x * 64 + wave * 16;

  const __bf16* Qb = qkv + (size_t)(b * Nseq + qrow) * ROWS + h * 64;
  const __bf16* Kb = qkv + (size_t)(b * Nseq) * ROWS + 1024 + h * 64;
  const __bf16* Vb = qkv + (size_t)(b * Nseq) * ROWS + 2048 + h * 64;

  v16bf qf[2];
  qf[0] = load_frag(Qb,      ROWS);
  qf[1] = load_frag(Qb + 32, ROWS);

  __shared__ __bf16 Vt[64][72];       // V tile transposed: [hd][key]
  __shared__ __bf16 Pl[4][16][72];    // per-wave probabilities
#if HAVE_TDM
  __shared__ __bf16 Kl[2][64][72];    // K tiles (TDM staged, double-buffered)
  const unsigned offK0 = (unsigned)(uintptr_t)(&Kl[0][0][0]);
  const unsigned offK1 = (unsigned)(uintptr_t)(&Kl[1][0][0]);
  if (wave == 0)
    tdm_load_tile(offK0, Kb, 64, 64, 64, (unsigned)ROWS, 4, 3);
#endif

  v8f o[4] = {};
  float m_i[8], l_i[8];
#pragma unroll
  for (int r = 0; r < 8; ++r) { m_i[r] = -3.0e30f; l_i[r] = 0.0f; }

  const float scale = 0.125f;         // 1/sqrt(64)
  const int rbase = (lane >> 4) * 8;
  const int nloc  = lane & 15;
  const int ntiles = Nseq / 64;

  for (int it = 0; it < ntiles; ++it) {
    const int kv0 = it * 64;
#if HAVE_TDM
    const int cur = it & 1;
    if (wave == 0) __builtin_amdgcn_s_wait_tensorcnt(0);
#endif
    __syncthreads();                  // prev-iter readers done; K tile landed
#if HAVE_TDM
    if (wave == 0 && (it + 1) < ntiles)
      tdm_load_tile(cur ? offK0 : offK1, Kb + (size_t)(kv0 + 64) * ROWS,
                    64, 64, 64, (unsigned)ROWS, 4, 3);
#endif

    // Stage V tile transposed: thread = 4 keys x 8 hd, packed b64 stores.
    {
      const int key0 = (tid & 15) * 4;
      const int d0   = (tid >> 4) * 8;
      union { uint4 q; __bf16 e[8]; } rowv[4];
#pragma unroll
      for (int kk = 0; kk < 4; ++kk)
        rowv[kk].q = *(const uint4*)(Vb + (size_t)(kv0 + key0 + kk) * ROWS + d0);
      if (kv0 + 64 < Nseq)
        __builtin_prefetch(Vb + (size_t)(kv0 + 64 + key0) * ROWS + d0, 0, 0);
#pragma unroll
      for (int dd = 0; dd < 8; ++dd) {
        union { __bf16 e[4]; unsigned long long q; } pk;
#pragma unroll
        for (int kk = 0; kk < 4; ++kk) pk.e[kk] = rowv[kk].e[dd];
        *(unsigned long long*)(&Vt[d0 + dd][key0]) = pk.q;
      }
    }

    // S = Q K^T
    v8f s[4];
#pragma unroll
    for (int j = 0; j < 4; ++j) {
#if HAVE_TDM
      v16bf k0 = load_frag(&Kl[cur][j * 16][0],  72);
      v16bf k1 = load_frag(&Kl[cur][j * 16][32], 72);
#else
      const __bf16* kb = Kb + (size_t)(kv0 + j * 16) * ROWS;
      v16bf k0 = load_frag(kb,      ROWS);
      v16bf k1 = load_frag(kb + 32, ROWS);
#endif
      v8f z = {};
      z = wmma_bf16(qf[0], k0, z);
      z = wmma_bf16(qf[1], k1, z);
      s[j] = z;
    }

    // Online softmax (reductions inside 16-lane rows -> permlane16).
#pragma unroll
    for (int r = 0; r < 8; ++r) {
      float mx = -3.0e30f;
#pragma unroll
      for (int j = 0; j < 4; ++j) {
        s[j][r] *= scale;
        mx = fmaxf(mx, s[j][r]);
      }
      mx = row16_max(mx);
      const float mn = fmaxf(m_i[r], mx);
      const float corr = __expf(m_i[r] - mn);
      m_i[r] = mn;
      l_i[r] *= corr;
#pragma unroll
      for (int j = 0; j < 4; ++j) o[j][r] *= corr;
      float rsum = 0.f;
#pragma unroll
      for (int j = 0; j < 4; ++j) {
        const float p = __expf(s[j][r] - mn);
        s[j][r] = p;
        rsum += p;
      }
      l_i[r] += row16_sum(rsum);
    }

    // Spill P (bf16) to this wave's LDS tile for the P*V WMMA.
#pragma unroll
    for (int j = 0; j < 4; ++j)
#pragma unroll
      for (int r = 0; r < 8; ++r)
        Pl[wave][rbase + r][j * 16 + nloc] = f2bf(s[j][r]);

    __syncthreads();                  // Vt fully staged; P in-order per wave

    // O += P * V   (K=64 keys -> 2 steps; N=hd -> 4 tiles of 16)
#pragma unroll
    for (int step = 0; step < 2; ++step) {
      v16bf pf = load_frag(&Pl[wave][0][step * 32], 72);
#pragma unroll
      for (int f = 0; f < 4; ++f) {
        v16bf vf = load_frag(&Vt[f * 16][step * 32], 72);
        o[f] = wmma_bf16(pf, vf, o[f]);
      }
    }
  }

#pragma unroll
  for (int f = 0; f < 4; ++f)
#pragma unroll
    for (int r = 0; r < 8; ++r) {
      const int row = b * Nseq + qrow + rbase + r;
      const int col = h * 64 + f * 16 + nloc;
      out[(size_t)row * 1024 + col] = f2bf(o[f][r] / l_i[r]);
    }
}

// ---------------------------------------------------------------------------
// Host-side orchestration.
// ---------------------------------------------------------------------------
extern "C" void kernel_launch(void* const* d_in, const int* in_sizes, int n_in,
                              void* d_out, int out_size, void* d_ws, size_t ws_size,
                              hipStream_t stream) {
  (void)in_sizes; (void)n_in; (void)out_size; (void)ws_size;
  const int D = 1024, HID = 4096, NSEQ = 2048, B = 4;
  const int M = B * NSEQ;

  const float* x      = (const float*)d_in[0];
  const float* ln1_w  = (const float*)d_in[1];
  const float* ln1_b  = (const float*)d_in[2];
  const float* qkv_w  = (const float*)d_in[3];
  const float* qkv_b  = (const float*)d_in[4];
  const float* proj_w = (const float*)d_in[5];
  const float* proj_b = (const float*)d_in[6];
  const float* ln2_w  = (const float*)d_in[7];
  const float* ln2_b  = (const float*)d_in[8];
  const float* fc1_w  = (const float*)d_in[9];
  const float* fc1_b  = (const float*)d_in[10];
  const float* fc2_w  = (const float*)d_in[11];
  const float* fc2_b  = (const float*)d_in[12];
  float* out = (float*)d_out;

  uintptr_t p = (uintptr_t)d_ws;
  auto carve = [&](size_t bytes) {
    void* r = (void*)p;
    p += (bytes + 255) & ~(size_t)255;
    return r;
  };
  __bf16* qkv_wt = (__bf16*)carve((size_t)3 * D * D * 2);
  __bf16* proj_wt= (__bf16*)carve((size_t)D * D * 2);
  __bf16* fc1_wt = (__bf16*)carve((size_t)HID * D * 2);
  __bf16* fc2_wt = (__bf16*)carve((size_t)D * HID * 2);
  __bf16* ln1o   = (__bf16*)carve((size_t)M * D * 2);
  __bf16* qkvo   = (__bf16*)carve((size_t)M * 3 * D * 2);
  __bf16* attno  = (__bf16*)carve((size_t)M * D * 2);
  float*  x1     = (float*) carve((size_t)M * D * 4);
  __bf16* h2     = (__bf16*)carve((size_t)M * D * 2);
  __bf16* fc1o   = (__bf16*)carve((size_t)M * HID * 2);

  transpose_cast_kernel<<<dim3(D / 32, 3 * D / 32), 256, 0, stream>>>(qkv_w,  qkv_wt,  D, 3 * D);
  transpose_cast_kernel<<<dim3(D / 32, D / 32),     256, 0, stream>>>(proj_w, proj_wt, D, D);
  transpose_cast_kernel<<<dim3(D / 32, HID / 32),   256, 0, stream>>>(fc1_w,  fc1_wt,  D, HID);
  transpose_cast_kernel<<<dim3(HID / 32, D / 32),   256, 0, stream>>>(fc2_w,  fc2_wt,  HID, D);

  layernorm_bf16_kernel<<<M, 256, 0, stream>>>(x, ln1_w, ln1_b, ln1o, D);

  gemm_wmma_kernel<0, false, true><<<dim3(M / 128, 3 * D / 128), 128, 0, stream>>>(
      ln1o, qkv_wt, qkv_b, nullptr, qkvo, M, 3 * D, D);

  flash_attn_kernel<<<dim3(NSEQ / 64, B * 16), 128, 0, stream>>>(qkvo, attno, NSEQ);

  gemm_wmma_kernel<0, true, false><<<dim3(M / 128, D / 128), 128, 0, stream>>>(
      attno, proj_wt, proj_b, x, (void*)x1, M, D, D);

  layernorm_bf16_kernel<<<M, 256, 0, stream>>>(x1, ln2_w, ln2_b, h2, D);

  gemm_wmma_kernel<1, false, true><<<dim3(M / 128, HID / 128), 128, 0, stream>>>(
      h2, fc1_wt, fc1_b, nullptr, fc1o, M, HID, D);

  gemm_wmma_kernel<0, true, false><<<dim3(M / 128, D / 128), 128, 0, stream>>>(
      fc1o, fc2_wt, fc2_b, x1, (void*)out, M, D, HID);
}
